// MultiHeadAttention_59047210385779
// MI455X (gfx1250) — compile-verified, matrix operand order checked by reading, not run on hardware
//
#include <hip/hip_runtime.h>
#include <hip/hip_bf16.h>

typedef __bf16 bf16_t;
typedef __attribute__((ext_vector_type(16))) __bf16 v16bf;
typedef __attribute__((ext_vector_type(8)))  float  v8f;
typedef __attribute__((ext_vector_type(4)))  float  v4f;
typedef unsigned int u32x4 __attribute__((ext_vector_type(4)));
typedef int          i32x4 __attribute__((ext_vector_type(4)));
typedef int          i32x8 __attribute__((ext_vector_type(8)));
// GCC-style vector type matching the async-store builtin's expected pointee
typedef int v4i_vs __attribute__((vector_size(4 * sizeof(int))));
typedef __attribute__((address_space(1))) v4i_vs* g_v4i_ptr;
typedef __attribute__((address_space(3))) v4i_vs* l_v4i_ptr;

#define B_  2
#define S_  2048
#define D_  1024
#define H_  16
#define DK_ 64
#define BSD ((size_t)B_ * S_ * D_)   // 4,194,304

#if __has_builtin(__builtin_amdgcn_tensor_load_to_lds)
#define HAVE_TDM 1
#else
#define HAVE_TDM 0
#endif

#if __has_builtin(__builtin_amdgcn_global_store_async_from_lds_b128)
#define HAVE_ASYNC_ST 1
#else
#define HAVE_ASYNC_ST 0
#endif

// K-index pattern for 16-bit A/B WMMA fragments (per ISA 7.12.2):
// lane half kh selects +8; elements 8..15 select +16.
static __device__ __forceinline__ int kpat(int e, int kh) {
    return (e & 7) + 8 * kh + ((e & 8) ? 16 : 0);
}

static __device__ __forceinline__ v8f wmma_bf16(v16bf a, v16bf b, v8f c) {
    return __builtin_amdgcn_wmma_f32_16x16x32_bf16(false, a, false, b, (short)0, c, false, false);
}

static __device__ __forceinline__ void cvt4(v16bf& d, int base, float4 f) {
    d[base + 0] = (bf16_t)f.x; d[base + 1] = (bf16_t)f.y;
    d[base + 2] = (bf16_t)f.z; d[base + 3] = (bf16_t)f.w;
}

static __device__ __forceinline__ void cvt4v(v16bf& d, int base, v4f f) {
    d[base + 0] = (bf16_t)f.x; d[base + 1] = (bf16_t)f.y;
    d[base + 2] = (bf16_t)f.z; d[base + 3] = (bf16_t)f.w;
}

// LDS byte offset of a generic pointer into shared memory (addrspacecast -> ptrtoint)
static __device__ __forceinline__ unsigned lds_offset_of(const void* p) {
    return (unsigned)(unsigned long long)
        (__attribute__((address_space(3))) const void*)p;
}

#if HAVE_TDM
// TDM: DMA one 64-key x 64-dk bf16 tile of the per-head V tensor [S_, DK_] into LDS.
static __device__ __forceinline__ void tdm_load_v_chunk(const bf16_t* vh, int kc, bf16_t* dst) {
    const unsigned long long gaddr =
        (unsigned long long)(const void*)vh +
        (unsigned long long)kc * 64 * DK_ * sizeof(bf16_t);
    u32x4 g0 = {};
    i32x8 g1 = {};
    i32x4 gz4 = {};
    // D# group 0: count=1 | lds_addr | 57-bit global_addr | type=2
    g0[0] = 1u;
    g0[1] = lds_offset_of(dst);
    g0[2] = (unsigned)(gaddr & 0xFFFFFFFFull);
    g0[3] = (unsigned)((gaddr >> 32) & 0x01FFFFFFull) | (2u << 30);
    // D# group 1: mask=0, data_size=2B, tensor [2048 x 64] bf16, tile 64x64,
    // dim0 stride = 64 elements.
    g1[0] = (1 << 16);                 // data_size = 2 bytes
    g1[1] = (DK_ & 0xFFFF) << 16;      // tensor_dim0[15:0]   (bits 79:48)
    g1[2] = (S_ & 0xFFFF) << 16;       // tensor_dim0 hi=0 | tensor_dim1[15:0]
    g1[3] = (64 << 16);                // tensor_dim1 hi=0 | tile_dim0=64
    g1[4] = 64;                        // tile_dim1=64 | tile_dim2=0
    g1[5] = DK_;                       // tensor_dim0_stride lo32
    g1[6] = 0;
    g1[7] = 0;
#if __clang_major__ >= 23
    i32x8 gz8 = {};
    __builtin_amdgcn_tensor_load_to_lds(g0, g1, gz4, gz4, gz8, 0);
#else
    __builtin_amdgcn_tensor_load_to_lds(g0, g1, gz4, gz4, 0);
#endif
}
#endif

// ---------------------------------------------------------------------------
// Kernel 1: fused QKV projection.  X @ W^T + b  ->  bf16 [B,H,S,DK] in ws.
// One wave per 16x16 output tile. grid = (D/16, B*S/16, 3)
// ---------------------------------------------------------------------------
__global__ void qkv_proj_kernel(const float* __restrict__ Q, const float* __restrict__ K,
                                const float* __restrict__ V,
                                const float* __restrict__ Wq, const float* __restrict__ bq,
                                const float* __restrict__ Wk, const float* __restrict__ bk,
                                const float* __restrict__ Wv, const float* __restrict__ bv,
                                bf16_t* __restrict__ qkv_ws) {
    const int lane  = threadIdx.x & 31;
    const int col15 = lane & 15;
    const int kh    = (lane >> 4) & 1;
    const int tileN = blockIdx.x;     // 0..63
    const int tileM = blockIdx.y;     // 0..255
    const int which = blockIdx.z;     // 0=q 1=k 2=v

    const float* X    = (which == 0) ? Q  : (which == 1) ? K  : V;
    const float* W    = (which == 0) ? Wq : (which == 1) ? Wk : Wv;
    const float* bias = (which == 0) ? bq : (which == 1) ? bk : bv;
    bf16_t* out = qkv_ws + (size_t)which * BSD;

    const int row_g = tileM * 16 + col15;      // A: lane = row
    const int col_g = tileN * 16 + col15;      // B: lane = col = W row

    const float* arow = X + (size_t)row_g * D_ + 8 * kh;
    const float* brow = W + (size_t)col_g * D_ + 8 * kh;

    v8f acc = {};
    for (int kb = 0; kb < D_; kb += 32) {
        __builtin_prefetch(arow + kb + 128, 0, 0);
        __builtin_prefetch(brow + kb + 128, 0, 0);
        const float4* ap = (const float4*)(arow + kb);
        const float4* bp = (const float4*)(brow + kb);
        float4 a0 = ap[0], a1 = ap[1], a2 = ap[4], a3 = ap[5];
        float4 b0 = bp[0], b1 = bp[1], b2 = bp[4], b3 = bp[5];
        v16bf a, b;
        cvt4(a, 0, a0); cvt4(a, 4, a1); cvt4(a, 8, a2); cvt4(a, 12, a3);
        cvt4(b, 0, b0); cvt4(b, 4, b1); cvt4(b, 8, b2); cvt4(b, 12, b3);
        acc = wmma_bf16(a, b, acc);
    }

    const float biasv = bias[col_g];
    // scatter to head-split layout [B,H,S,DK]
    const int h    = tileN >> 2;
    const int dk0  = (tileN & 3) * 16;
    const int bidx = tileM >> 7;
    const int s0   = (tileM & 127) * 16;
    bf16_t* obase = out + (size_t)(bidx * H_ + h) * S_ * DK_;
    #pragma unroll
    for (int r = 0; r < 8; ++r) {
        const int m = r + 8 * kh;
        obase[(size_t)(s0 + m) * DK_ + dk0 + col15] = (bf16_t)(acc[r] + biasv);
    }
}

// ---------------------------------------------------------------------------
// Kernel 2: per-(b,h,q-tile) attention. 4 waves / block, ~145 KB dynamic LDS.
// Phase A: scores = (q k^T)/8 + causal mask   (WMMA)
// Phase B: row softmax (vectorized), normalize in LDS, async-store attn
// Phase C: out_tile = P @ V  (WMMA; V double-buffered in LDS via the TDM)
// grid = (S/16, H, B), block = 128
// ---------------------------------------------------------------------------
__global__ void attn_kernel(const bf16_t* __restrict__ qkv_ws,
                            float* __restrict__ attn_out,
                            bf16_t* __restrict__ ao_bf16) {
    extern __shared__ float smem[];
    float*  sc  = smem;                  // [16][S_] scores -> probabilities
    float*  red = smem + 16 * S_;        // [16][8]  reduction scratch
    bf16_t* sV0 = (bf16_t*)(red + 16 * 8);       // [64][DK_] V chunk buf 0 (8 KB)
    bf16_t* sV1 = sV0 + 64 * DK_;                // [64][DK_] V chunk buf 1 (8 KB)

    const int tid   = threadIdx.x;
    const int lane  = tid & 31;
    const int wv    = tid >> 5;          // wave 0..3
    const int col15 = lane & 15;
    const int kh    = (lane >> 4) & 1;

    const int qt = blockIdx.x;           // q tile 0..127
    const int h  = blockIdx.y;
    const int b  = blockIdx.z;
    const int q0 = qt * 16;

    const size_t headOff = (size_t)(b * H_ + h) * S_ * DK_;
    const bf16_t* qh   = qkv_ws + headOff;
    const bf16_t* kh_p = qkv_ws + BSD + headOff;
    const bf16_t* vh   = qkv_ws + 2 * BSD + headOff;

    // q A-fragments for DK chunks [0,32) and [32,64): lane = q row
    const bf16_t* qrow = qh + (size_t)(q0 + col15) * DK_ + 8 * kh;
    v16bf aq0, aq1;
    #pragma unroll
    for (int e = 0; e < 8; ++e) {
        aq0[e] = qrow[e];      aq0[8 + e] = qrow[16 + e];
        aq1[e] = qrow[32 + e]; aq1[8 + e] = qrow[48 + e];
    }

    // ---- Phase A: scores into LDS ----
    for (int kt = wv; kt < S_ / 16; kt += 4) {
        const int n = kt * 16 + col15;   // key position (B: lane = col)
        const bf16_t* krow = kh_p + (size_t)n * DK_ + 8 * kh;
        v16bf bk0, bk1;
        #pragma unroll
        for (int e = 0; e < 8; ++e) {
            bk0[e] = krow[e];      bk0[8 + e] = krow[16 + e];
            bk1[e] = krow[32 + e]; bk1[8 + e] = krow[48 + e];
        }
        v8f acc = {};
        acc = wmma_bf16(aq0, bk0, acc);
        acc = wmma_bf16(aq1, bk1, acc);
        const int mbase = 8 * kh;
        #pragma unroll
        for (int r = 0; r < 8; ++r) {
            const int m = r + mbase;
            float sv = acc[r] * 0.125f;          // 1/sqrt(64)
            if (n > q0 + m) sv = -1e9f;          // causal mask
            sc[m * S_ + n] = sv;
        }
    }
    __syncthreads();

    // ---- Phase B: softmax (8 threads per row, 256 cols each, float4 LDS ops) ----
    const int r  = tid >> 3;
    const int j  = tid & 7;
    const int c0 = j * (S_ / 8);
    v4f* scr = (v4f*)(sc + r * S_ + c0);         // 64 vectors per slice
    float mx = -3.0e38f;
    for (int i = 0; i < S_ / 32; ++i) {
        v4f v = scr[i];
        mx = fmaxf(fmaxf(mx, v.x), fmaxf(fmaxf(v.y, v.z), v.w));
    }
    red[r * 8 + j] = mx;
    __syncthreads();
    float rowmax = red[r * 8];
    #pragma unroll
    for (int t = 1; t < 8; ++t) rowmax = fmaxf(rowmax, red[r * 8 + t]);
    __syncthreads();                     // done reading red
    float ssum = 0.f;
    for (int i = 0; i < S_ / 32; ++i) {
        v4f v = scr[i];
        v.x = __expf(v.x - rowmax); v.y = __expf(v.y - rowmax);
        v.z = __expf(v.z - rowmax); v.w = __expf(v.w - rowmax);
        scr[i] = v;
        ssum += (v.x + v.y) + (v.z + v.w);
    }
    red[r * 8 + j] = ssum;
    __syncthreads();
    float rowsum = 0.f;
    #pragma unroll
    for (int t = 0; t < 8; ++t) rowsum += red[r * 8 + t];
    const float iv = 1.0f / rowsum;
    // normalize probabilities IN LDS (Phase C consumes them directly)
    for (int i = 0; i < S_ / 32; ++i) scr[i] = scr[i] * iv;

    // stream attn probabilities to global: async LDS->global if available
    // (drains in the background during Phase C WMMAs), else NT stores.
    float* aout = attn_out + ((size_t)(b * H_ + h) * S_ + (q0 + r)) * S_ + c0;
#if HAVE_ASYNC_ST
    for (int i = 0; i < S_ / 32; ++i)
        __builtin_amdgcn_global_store_async_from_lds_b128(
            (g_v4i_ptr)((v4f*)aout + i), (l_v4i_ptr)(scr + i), 0, 0);
#else
    for (int i = 0; i < S_ / 32; ++i)
        __builtin_nontemporal_store(scr[i], (v4f*)aout + i);
#endif

    // ---- Phase C: out_tile = P @ V ; wave wv owns dk columns [wv*16, wv*16+16) ----
    // V staged in 64x64 bf16 chunks, double-buffered, DMA'd by the TDM and
    // overlapped with WMMA compute (tensor ops complete in order per wave).
    const int nt = wv;
    const int n  = nt * 16 + col15;      // dk column (B: lane = col)
    v8f acc = {};
#if HAVE_TDM
    if (wv == 0) tdm_load_v_chunk(vh, 0, sV0);
#endif
    for (int kc = 0; kc < S_ / 64; ++kc) {
        bf16_t* cur = (kc & 1) ? sV1 : sV0;
#if HAVE_TDM
        if (wv == 0) {
            if (kc + 1 < S_ / 64) {
                tdm_load_v_chunk(vh, kc + 1, (kc & 1) ? sV0 : sV1);  // prefetch next
                __builtin_amdgcn_s_wait_tensorcnt(1);                // current resident
            } else {
                __builtin_amdgcn_s_wait_tensorcnt(0);
            }
        }
#else
        {   // fallback: cooperative 8 KB copy
            const uint4* srcv = (const uint4*)(vh + (size_t)kc * 64 * DK_);
            uint4* dstv = (uint4*)cur;
            for (int i = tid; i < (64 * DK_) / 8; i += 128) dstv[i] = srcv[i];
        }
#endif
        __syncthreads();                 // chunk visible to all waves
        #pragma unroll
        for (int half = 0; half < 2; ++half) {
            // A: lane = q row; two contiguous 8-float groups per lane
            const float* pbase = sc + col15 * S_ + kc * 64 + half * 32 + 8 * kh;
            v4f p0 = *(const v4f*)(pbase + 0);
            v4f p1 = *(const v4f*)(pbase + 4);
            v4f p2 = *(const v4f*)(pbase + 16);
            v4f p3 = *(const v4f*)(pbase + 20);
            v16bf pa, vb;
            cvt4v(pa, 0, p0); cvt4v(pa, 4, p1); cvt4v(pa, 8, p2); cvt4v(pa, 12, p3);
            #pragma unroll
            for (int e = 0; e < 16; ++e)
                vb[e] = cur[(half * 32 + kpat(e, kh)) * DK_ + n];
            acc = wmma_bf16(pa, vb, acc);
        }
        __syncthreads();                 // all reads of cur done before it is reused
    }
    const int mbase = 8 * kh;
    #pragma unroll
    for (int r2 = 0; r2 < 8; ++r2) {
        const int m = r2 + mbase;
        ao_bf16[((size_t)(b * S_ + q0 + m)) * D_ + h * DK_ + nt * 16 + col15] =
            (bf16_t)acc[r2];
    }
#if HAVE_ASYNC_ST
#if __has_builtin(__builtin_amdgcn_s_wait_asynccnt)
    __builtin_amdgcn_s_wait_asynccnt(0);
#else
    asm volatile("s_wait_asynccnt 0x0" ::: "memory");
#endif
#endif
}

// ---------------------------------------------------------------------------
// Kernel 3: output projection. AO(bf16) @ Wo^T + bo -> fp32 out.
// grid = (D/16, B*S/16), one wave per tile.
// ---------------------------------------------------------------------------
__global__ void out_proj_kernel(const bf16_t* __restrict__ AO,
                                const float* __restrict__ Wo,
                                const float* __restrict__ bo,
                                float* __restrict__ out) {
    const int lane  = threadIdx.x & 31;
    const int col15 = lane & 15;
    const int kh    = (lane >> 4) & 1;
    const int tileN = blockIdx.x;
    const int tileM = blockIdx.y;
    const int row_g = tileM * 16 + col15;
    const int col_g = tileN * 16 + col15;

    const bf16_t* arow = AO + (size_t)row_g * D_ + 8 * kh;
    const float*  brow = Wo + (size_t)col_g * D_ + 8 * kh;

    v8f acc = {};
    for (int kb = 0; kb < D_; kb += 32) {
        __builtin_prefetch(brow + kb + 128, 0, 0);
        v16bf a, bfr;
        const bf16_t* ap = arow + kb;
        #pragma unroll
        for (int e = 0; e < 8; ++e) { a[e] = ap[e]; a[8 + e] = ap[16 + e]; }
        const float4* bp = (const float4*)(brow + kb);
        float4 b0 = bp[0], b1 = bp[1], b2 = bp[4], b3 = bp[5];
        cvt4(bfr, 0, b0); cvt4(bfr, 4, b1); cvt4(bfr, 8, b2); cvt4(bfr, 12, b3);
        acc = wmma_bf16(a, bfr, acc);
    }
    const float bb = bo[col_g];
    #pragma unroll
    for (int r = 0; r < 8; ++r) {
        out[(size_t)(tileM * 16 + r + 8 * kh) * D_ + col_g] = acc[r] + bb;
    }
}

// ---------------------------------------------------------------------------
extern "C" void kernel_launch(void* const* d_in, const int* in_sizes, int n_in,
                              void* d_out, int out_size, void* d_ws, size_t ws_size,
                              hipStream_t stream) {
    (void)in_sizes; (void)n_in; (void)out_size; (void)ws_size;
    const float* Q  = (const float*)d_in[0];
    const float* K  = (const float*)d_in[1];
    const float* V  = (const float*)d_in[2];
    // d_in[3] = mask (int32 tril) -- causal mask applied analytically
    const float* Wq = (const float*)d_in[4];
    const float* bq = (const float*)d_in[5];
    const float* Wk = (const float*)d_in[6];
    const float* bk = (const float*)d_in[7];
    const float* Wv = (const float*)d_in[8];
    const float* bv = (const float*)d_in[9];
    const float* Wo = (const float*)d_in[10];
    const float* bo = (const float*)d_in[11];

    bf16_t* qkv = (bf16_t*)d_ws;               // 3 * BSD bf16 (q,k,v head-split)
    bf16_t* ao  = qkv + 3 * BSD;               // BSD bf16 (pre-projection attn out)
    float*  out  = (float*)d_out;              // [B,S,D] fp32
    float*  attn = out + BSD;                  // [B,H,S,S] fp32

    // 1) QKV projections
    qkv_proj_kernel<<<dim3(D_ / 16, (B_ * S_) / 16, 3), 32, 0, stream>>>(
        Q, K, V, Wq, bq, Wk, bk, Wv, bv, qkv);

    // 2) attention (128 KB scores + 2x8 KB V double-buffer; WGP has 320 KB)
    const size_t smemBytes =
        (size_t)(16 * S_ + 16 * 8) * sizeof(float) + (size_t)2 * 64 * DK_ * sizeof(bf16_t);
    (void)hipFuncSetAttribute((const void*)attn_kernel,
                              hipFuncAttributeMaxDynamicSharedMemorySize, (int)smemBytes);
    attn_kernel<<<dim3(S_ / 16, H_, B_), 128, smemBytes, stream>>>(qkv, attn, ao);

    // 3) output projection
    out_proj_kernel<<<dim3(D_ / 16, (B_ * S_) / 16), 32, 0, stream>>>(ao, Wo, bo, out);
}
